// EventADModel_25640954757324
// MI455X (gfx1250) — compile-verified, matrix operand order checked by reading, not run on hardware
//
#include <hip/hip_runtime.h>
#include <hip/hip_bf16.h>

typedef __attribute__((ext_vector_type(16))) _Float16 v16h;
typedef __attribute__((ext_vector_type(8)))  _Float16 v8h;
typedef __attribute__((ext_vector_type(8)))  float    v8f;

#define DEV __device__ __forceinline__

static constexpr int TB = 30;        // time steps
static constexpr int NB = 8192 * 30; // total rows
static constexpr int PW = 10752;     // per-wave LDS halfs

DEV v8f wmma16(v16h a, v16h b, v8f c) {
  return __builtin_amdgcn_wmma_f32_16x16x32_f16(false, a, false, b, (short)0, c,
                                                false, false);
}

// Build a 16x16x32 WMMA fragment (A or B operand) for this lane from a
// row-major f16 panel row pointer p (already offset to [row][kbase]).
// Lane group g selects K runs {8g..8g+7} and {16+8g..23+8g}: two b128 loads.
DEV v16h make_frag(const _Float16* p, int g) {
  v8h lo = *(const v8h*)(p + 8 * g);
  v8h hi = *(const v8h*)(p + 16 + 8 * g);
  v16h r;
#pragma unroll
  for (int i = 0; i < 8; ++i) { r[i] = lo[i]; r[i + 8] = hi[i]; }
  return r;
}

DEV v8f zero8() { v8f z = {0.f,0.f,0.f,0.f,0.f,0.f,0.f,0.f}; return z; }

// Branch-free, division-free activations on v_exp_f32 + v_rcp_f32.
DEV float sigmf(float x) {
  return __builtin_amdgcn_rcpf(1.f + __expf(-x));
}
DEV float tanh_fast(float x) {
  // tanh(x) = 1 - 2/(exp(2x)+1); exact at +/-inf via rcp(inf)=0, rcp(1)=1.
  return 1.f - 2.f * __builtin_amdgcn_rcpf(1.f + __expf(2.f * x));
}

// One GRU step for hidden size 256, 16 rows per wave.
// xpan: [16][KXT*32] f16 LDS panel, hpan: [16][256] f16 LDS panel (in/out).
// Wih packed f16 row-major [768][KXT*32], Whh packed [768][256].
template<int KXT, bool HASH>
DEV void gru_layer256(const _Float16* xpan, _Float16* hpan,
                      const _Float16* __restrict__ Wih,
                      const _Float16* __restrict__ Whh,
                      const float* __restrict__ bih,
                      const float* __restrict__ bhh, int lane) {
  constexpr int IX = KXT * 32;
  const int m = lane & 15, g = lane >> 4, n = m;
  v16h ax[KXT];
#pragma unroll
  for (int k = 0; k < KXT; ++k) ax[k] = make_frag(xpan + m * IX + k * 32, g);
  v16h ah[8];
  if (HASH) {
#pragma unroll
    for (int k = 0; k < 8; ++k) ah[k] = make_frag(hpan + m * 256 + k * 32, g);
  }
#pragma unroll 1
  for (int jc = 0; jc < 16; ++jc) {
    const int col = jc * 16 + n;
    v8f ir = zero8(), iz = zero8(), in_ = zero8();
    v8f hr = zero8(), hz = zero8(), hn = zero8();
#pragma unroll
    for (int k = 0; k < KXT; ++k) {
      ir  = wmma16(ax[k], make_frag(Wih + (size_t)(      col) * IX + k * 32, g), ir);
      iz  = wmma16(ax[k], make_frag(Wih + (size_t)(256 + col) * IX + k * 32, g), iz);
      in_ = wmma16(ax[k], make_frag(Wih + (size_t)(512 + col) * IX + k * 32, g), in_);
    }
    if (HASH) {
#pragma unroll
      for (int k = 0; k < 8; ++k) {
        hr = wmma16(ah[k], make_frag(Whh + (size_t)(      col) * 256 + k * 32, g), hr);
        hz = wmma16(ah[k], make_frag(Whh + (size_t)(256 + col) * 256 + k * 32, g), hz);
        hn = wmma16(ah[k], make_frag(Whh + (size_t)(512 + col) * 256 + k * 32, g), hn);
      }
    }
    const float bir = bih[col], biz = bih[256 + col], bin = bih[512 + col];
    const float bhr = bhh[col], bhz = bhh[256 + col], bhn = bhh[512 + col];
#pragma unroll
    for (int r = 0; r < 8; ++r) {
      float hold = HASH ? (float)hpan[(g * 8 + r) * 256 + col] : 0.f;
      float rg = sigmf(ir[r] + bir + hr[r] + bhr);
      float zg = sigmf(iz[r] + biz + hz[r] + bhz);
      float ng = tanh_fast(in_[r] + bin + rg * (hn[r] + bhn));
      hpan[(g * 8 + r) * 256 + col] = (_Float16)((1.f - zg) * ng + zg * hold);
    }
  }
}

// Prologue: f32 weights -> packed row-major f16 with optional O/I padding.
__global__ void pack_w_f16(const float* __restrict__ src, _Float16* __restrict__ dst,
                           int Osrc, int I, int Ipad, int total) {
  int idx = blockIdx.x * 256 + threadIdx.x;
  if (idx >= total) return;
  int o = idx / Ipad, i = idx - o * Ipad;
  float v = (o < Osrc && i < I) ? src[(size_t)o * I + i] : 0.f;
  dst[idx] = (_Float16)v;
}

__global__ __launch_bounds__(256, 1) void gru_fused_kernel(
    const float* __restrict__ feats, const float* __restrict__ coords,
    const unsigned char* __restrict__ mask,
    const _Float16* __restrict__ Wih0, const _Float16* __restrict__ Whh0,
    const float* __restrict__ bih0, const float* __restrict__ bhh0,
    const _Float16* __restrict__ Wih1, const _Float16* __restrict__ Whh1,
    const float* __restrict__ bih1, const float* __restrict__ bhh1,
    const _Float16* __restrict__ WihC, const float* __restrict__ bihC,
    const float* __restrict__ bhhC,
    const _Float16* __restrict__ We, const float* __restrict__ be,
    const _Float16* __restrict__ Wc, const float* __restrict__ bc,
    const _Float16* __restrict__ W1, const float* __restrict__ b1,
    const _Float16* __restrict__ W2, const float* __restrict__ b2,
    float* __restrict__ out) {
  extern __shared__ _Float16 smem[];
  const int lane = threadIdx.x & 31;
  const int wv = threadIdx.x >> 5;
  _Float16* base = smem + wv * PW;
  _Float16* xpan = base;           // [2][16][64]   (2048 halfs)
  _Float16* h0p  = base + 2048;    // [16][256]     (4096 halfs)
  _Float16* h1p  = base + 6144;    // [16][256]     (4096 halfs)
  _Float16* cpn  = base + 10240;   // [16][32]      (512 halfs)
  float*    scr  = (float*)cpn;    // [16][16] f32 overlay (after cpn is dead)

  const int r0 = (blockIdx.x * 8 + wv) * 16;

  // ---- stage x (both frames) as f16 [16][64] per frame -------------------
  for (int idx = lane; idx < 2 * 16 * 64; idx += 32) {
    int f = idx >> 10;
    int rem = idx & 1023;
    int row = rem >> 6, c = rem & 63;
    int nn = r0 + row;
    int b = nn / TB, t = nn - b * TB;
    xpan[idx] = (_Float16)feats[(((size_t)b * 2 + f) * TB + t) * 64 + c];
  }
  // ---- stage coords zero-padded to [16][32] -----------------------------
  for (int idx = lane; idx < 512; idx += 32) {
    int row = idx >> 5, c = idx & 31;
    float v = (c < 4) ? coords[(size_t)(r0 + row) * 4 + c] : 0.f;
    cpn[idx] = (_Float16)v;
  }
  __syncthreads();

  // ---- event GRU, 2 layers x 2 frames (h=0 at t=0 => gh = bhh) ----------
  gru_layer256<2, false>(xpan,        h0p, Wih0, Whh0, bih0, bhh0, lane);
  gru_layer256<8, false>(h0p,         h1p, Wih1, Whh1, bih1, bhh1, lane);
  gru_layer256<2, true >(xpan + 1024, h0p, Wih0, Whh0, bih0, bhh0, lane);
  gru_layer256<8, true >(h0p,         h1p, Wih1, Whh1, bih1, bhh1, lane);

  // ---- coord GRU: one step from h=0 (gh = bhhC, no Whh matmul) ----------
  {
    const int m = lane & 15, g = lane >> 4, n = m;
    v16h ac = make_frag(cpn + m * 32, g);
#pragma unroll
    for (int jc = 0; jc < 2; ++jc) {
      const int col = jc * 16 + n;
      v8f ir = zero8(), iz = zero8(), in_ = zero8();
      ir  = wmma16(ac, make_frag(WihC + (size_t)(     col) * 32, g), ir);
      iz  = wmma16(ac, make_frag(WihC + (size_t)(32 + col) * 32, g), iz);
      in_ = wmma16(ac, make_frag(WihC + (size_t)(64 + col) * 32, g), in_);
      float bir = bihC[col], biz = bihC[32 + col], bin = bihC[64 + col];
      float bhr = bhhC[col], bhz = bhhC[32 + col], bhn = bhhC[64 + col];
#pragma unroll
      for (int r = 0; r < 8; ++r) {
        float rg = sigmf(ir[r] + bir + bhr);
        float zg = sigmf(iz[r] + biz + bhz);
        float ng = tanh_fast(in_[r] + bin + rg * bhn);
        cpn[(g * 8 + r) * 32 + col] = (_Float16)((1.f - zg) * ng);
      }
    }
  }
  __syncthreads();

  // ---- fusion MLP --------------------------------------------------------
  {
    const int m = lane & 15, g = lane >> 4, n = m;
    // ep = h1 @ We.T + be  -> concat panel (h0p) cols [0..127]
    v16h a1[8];
#pragma unroll
    for (int k = 0; k < 8; ++k) a1[k] = make_frag(h1p + m * 256 + k * 32, g);
#pragma unroll 1
    for (int jc = 0; jc < 8; ++jc) {
      v8f acc = zero8();
#pragma unroll
      for (int k = 0; k < 8; ++k)
        acc = wmma16(a1[k], make_frag(We + (size_t)(jc * 16 + n) * 256 + k * 32, g), acc);
      float bb = be[jc * 16 + n];
#pragma unroll
      for (int r = 0; r < 8; ++r)
        h0p[(g * 8 + r) * 256 + jc * 16 + n] = (_Float16)(acc[r] + bb);
    }
    // cp = coord_h @ Wc.T + bc -> concat panel cols [128..255]
    v16h ach = make_frag(cpn + m * 32, g);
#pragma unroll 1
    for (int jc = 0; jc < 8; ++jc) {
      v8f acc = zero8();
      acc = wmma16(ach, make_frag(Wc + (size_t)(jc * 16 + n) * 32, g), acc);
      float bb = bc[jc * 16 + n];
#pragma unroll
      for (int r = 0; r < 8; ++r)
        h0p[(g * 8 + r) * 256 + 128 + jc * 16 + n] = (_Float16)(acc[r] + bb);
    }
    // hid = relu(concat @ W1.T + b1) -> hid panel (reuse xpan) [16][128]
    v16h afc[8];
#pragma unroll
    for (int k = 0; k < 8; ++k) afc[k] = make_frag(h0p + m * 256 + k * 32, g);
    _Float16* hidp = xpan;
#pragma unroll 1
    for (int jc = 0; jc < 8; ++jc) {
      v8f acc = zero8();
#pragma unroll
      for (int k = 0; k < 8; ++k)
        acc = wmma16(afc[k], make_frag(W1 + (size_t)(jc * 16 + n) * 256 + k * 32, g), acc);
      float bb = b1[jc * 16 + n];
#pragma unroll
      for (int r = 0; r < 8; ++r) {
        float v = acc[r] + bb;
        hidp[(g * 8 + r) * 128 + jc * 16 + n] = (_Float16)(v > 0.f ? v : 0.f);
      }
    }
    // logits = hid @ W2p.T + b2 (W2 zero-padded to [16][128]); cols 0,1 valid
    v16h ahd[4];
#pragma unroll
    for (int k = 0; k < 4; ++k) ahd[k] = make_frag(hidp + m * 128 + k * 32, g);
    v8f acc = zero8();
#pragma unroll
    for (int k = 0; k < 4; ++k)
      acc = wmma16(ahd[k], make_frag(W2 + (size_t)n * 128 + k * 32, g), acc);
    float bb = (n < 2) ? b2[n] : 0.f;
#pragma unroll
    for (int r = 0; r < 8; ++r) scr[(g * 8 + r) * 16 + n] = acc[r] + bb;
  }

  // ---- masked store: 16 rows x 2 logits = 32 values, one per lane -------
  {
    int row = lane >> 1, c = lane & 1;
    int nn = r0 + row;
    float v = scr[row * 16 + c];
    out[(size_t)nn * 2 + c] = mask[nn] ? v : 0.f;
  }
}

extern "C" void kernel_launch(void* const* d_in, const int* in_sizes, int n_in,
                              void* d_out, int out_size, void* d_ws, size_t ws_size,
                              hipStream_t stream) {
  (void)in_sizes; (void)n_in; (void)out_size; (void)ws_size;
  _Float16* ws = (_Float16*)d_ws;
  size_t off = 0;
  auto take = [&](size_t nh) { _Float16* p = ws + off; off += nh; return p; };
  _Float16* Wih0p = take(768 * 64);
  _Float16* Whh0p = take(768 * 256);
  _Float16* Wih1p = take(768 * 256);
  _Float16* Whh1p = take(768 * 256);
  _Float16* WihCp = take(96 * 32);     // I padded 4 -> 32
  _Float16* Wep   = take(128 * 256);
  _Float16* Wcp   = take(128 * 32);
  _Float16* W1p   = take(128 * 256);
  _Float16* W2p   = take(16 * 128);    // O padded 2 -> 16

  auto pack = [&](int si, _Float16* d, int Os, int I, int Op, int Ip) {
    int tot = Op * Ip;
    pack_w_f16<<<(tot + 255) / 256, 256, 0, stream>>>((const float*)d_in[si], d,
                                                      Os, I, Ip, tot);
  };
  pack(3,  Wih0p, 768, 64,  768, 64);
  pack(4,  Whh0p, 768, 256, 768, 256);
  pack(7,  Wih1p, 768, 256, 768, 256);
  pack(8,  Whh1p, 768, 256, 768, 256);
  pack(11, WihCp, 96,  4,   96,  32);
  pack(15, Wep,   128, 256, 128, 256);
  pack(17, Wcp,   128, 32,  128, 32);
  pack(19, W1p,   128, 256, 128, 256);
  pack(21, W2p,   2,   128, 16,  128);

  const size_t shbytes = (size_t)8 * PW * sizeof(_Float16); // 172,032 B
  (void)hipFuncSetAttribute(reinterpret_cast<const void*>(gru_fused_kernel),
                            hipFuncAttributeMaxDynamicSharedMemorySize,
                            (int)shbytes);
  gru_fused_kernel<<<NB / 128, 256, shbytes, stream>>>(
      (const float*)d_in[0], (const float*)d_in[1],
      (const unsigned char*)d_in[2],
      Wih0p, Whh0p, (const float*)d_in[5], (const float*)d_in[6],
      Wih1p, Whh1p, (const float*)d_in[9], (const float*)d_in[10],
      WihCp, (const float*)d_in[13], (const float*)d_in[14],
      Wep, (const float*)d_in[16], Wcp, (const float*)d_in[18],
      W1p, (const float*)d_in[20], W2p, (const float*)d_in[22],
      (float*)d_out);
}
